// GGNN_embedding_18021682774394
// MI455X (gfx1250) — matrix operand order repaired and checked by reading, not compiled, not run on hardware
//
#include <hip/hip_runtime.h>
#include <hip/hip_bf16.h>
#include <math.h>

#define MN 65536      // B*N nodes
#define BG 64         // graphs
#define NG 1024       // nodes per graph
#define FH 512
#define FIN0 256
#define EN 524288     // edges
#define KK1 717
#define KK2 502
#define KK3 352

typedef __attribute__((ext_vector_type(16))) __bf16 v16bf;
typedef __attribute__((ext_vector_type(8)))  float  v8f;

__device__ __forceinline__ unsigned short f2bf(float f) {
  unsigned int u = __float_as_uint(f);
  u += 0x7FFFu + ((u >> 16) & 1u);
  return (unsigned short)(u >> 16);
}

__device__ __forceinline__ float bf2f(unsigned int s16) {
  return __uint_as_float(s16 << 16);
}

__global__ void k_fill_mask(unsigned char* m, int n) {
  int i = blockIdx.x * blockDim.x + threadIdx.x;
  if (i < n) m[i] = 1;
}

__global__ void k_zero(float* a, int n) {
  int i = blockIdx.x * blockDim.x + threadIdx.x;
  if (i < n) a[i] = 0.f;
}

__global__ void k_f2bf(const float* __restrict__ in, unsigned short* __restrict__ out, int n) {
  int i = blockIdx.x * blockDim.x + threadIdx.x;
  if (i < n) out[i] = f2bf(in[i]);
}

// Pack W[K,N] (row-major fp32) into wave32 WMMA B-fragment order (bf16):
// packed[((kt*ntn+nt)*32+lane)*16 + e16], K=kt*32 + (e16>>3)*16 + (lane>>4)*8 + (e16&7).
__global__ void k_pack_w(const float* __restrict__ W, unsigned short* __restrict__ out,
                         int K, int N) {
  int p = blockIdx.x * blockDim.x + threadIdx.x;
  if (p >= K * N) return;
  int e16  = p & 15;
  int lane = (p >> 4) & 31;
  int tile = p >> 9;
  int ntn = N >> 4;
  int kt = tile / ntn;
  int nt = tile - kt * ntn;
  int j  = e16 >> 1;
  int h2 = e16 & 1;
  int kl = ((e16 >> 3) << 4) + ((lane >> 4) << 3) + (e16 & 7); (void)j; (void)h2;
  int n  = (nt << 4) + (lane & 15);
  int kg = (kt << 5) + kl;
  out[p] = f2bf(W[(size_t)kg * N + n]);
}

// C[M,N] = A[M,K](bf16) * Bpacked + bias.
// One wave computes a 16x64 strip: 4 fp32 accumulators share one A fragment per
// 32-deep K step -> 4x A reuse, 4 v_wmma_f32_16x16x32_bf16 per iteration.
// Output either fp32 (Cf) or bf16 (Cb) selected by bf16out.
__global__ void __launch_bounds__(256) k_gemm(const unsigned short* __restrict__ A,
                                              const unsigned short* __restrict__ Bp,
                                              const float* __restrict__ bias,
                                              float* __restrict__ Cf,
                                              unsigned short* __restrict__ Cb,
                                              int M, int K, int N, int bf16out) {
  int wave = threadIdx.x >> 5;
  int lane = threadIdx.x & 31;
  int tile = blockIdx.x * 8 + wave;
  int ntc = N >> 6;                    // 16x64 strips per tile-row (8)
  int tm = tile / ntc;
  int tc = tile - tm * ntc;
  int half = lane >> 4;
  int mrow = lane & 15;
  const unsigned short* arow  = A + (size_t)(tm * 16 + mrow) * K + half * 8;
  int ntn = N >> 4;
  size_t bstride = (size_t)ntn * 512;  // packed elements per 32-deep K step
  const unsigned short* bbase = Bp + ((size_t)(tc * 4) * 32 + lane) * 16;
  union U { v16bf v; unsigned int u[8]; };
  U a, b0, b1, b2, b3;
  v8f cacc[4];
#pragma unroll
  for (int j = 0; j < 4; ++j) cacc[j] = (v8f){0.f, 0.f, 0.f, 0.f, 0.f, 0.f, 0.f, 0.f};
  int ksteps = K >> 5;
  for (int kt = 0; kt < ksteps; ++kt) {
    // A layout (16-bit 16x32): dword j<4: K = half*8 + 2j ; j>=4: K = 16 + half*8 + 2(j-4)
    const unsigned int* ap = (const unsigned int*)(arow + (kt << 5));
    a.u[0] = ap[0];  a.u[1] = ap[1];  a.u[2] = ap[2];  a.u[3] = ap[3];
    a.u[4] = ap[8];  a.u[5] = ap[9];  a.u[6] = ap[10]; a.u[7] = ap[11];
    const unsigned int* bp = (const unsigned int*)(bbase + (size_t)kt * bstride);
    __builtin_prefetch(ap + 16, 0, 1);                    // next K-step A chunk
    __builtin_prefetch((const char*)bp + bstride * 2, 0, 1);  // next K-step B
#pragma unroll
    for (int i = 0; i < 8; ++i) b0.u[i] = bp[i];
#pragma unroll
    for (int i = 0; i < 8; ++i) b1.u[i] = bp[256 + i];    // +512 ushorts per nt
#pragma unroll
    for (int i = 0; i < 8; ++i) b2.u[i] = bp[512 + i];
#pragma unroll
    for (int i = 0; i < 8; ++i) b3.u[i] = bp[768 + i];
    cacc[0] = __builtin_amdgcn_wmma_f32_16x16x32_bf16(false, a.v, false, b0.v,
                                                      (short)0, cacc[0], false, false);
    cacc[1] = __builtin_amdgcn_wmma_f32_16x16x32_bf16(false, a.v, false, b1.v,
                                                      (short)0, cacc[1], false, false);
    cacc[2] = __builtin_amdgcn_wmma_f32_16x16x32_bf16(false, a.v, false, b2.v,
                                                      (short)0, cacc[2], false, false);
    cacc[3] = __builtin_amdgcn_wmma_f32_16x16x32_bf16(false, a.v, false, b3.v,
                                                      (short)0, cacc[3], false, false);
  }
  int m0 = tm * 16 + half * 8;
#pragma unroll
  for (int j = 0; j < 4; ++j) {
    int col = (tc * 4 + j) * 16 + mrow;
    float bs = bias[col];
    if (bf16out) {
      unsigned short* o = Cb + (size_t)m0 * N + col;
#pragma unroll
      for (int r = 0; r < 8; ++r) o[(size_t)r * N] = f2bf(cacc[j][r] + bs);
    } else {
      float* o = Cf + (size_t)m0 * N + col;
#pragma unroll
      for (int r = 0; r < 8; ++r) o[(size_t)r * N] = cacc[j][r] + bs;
    }
  }
}

// out[dst] += sigmoid(k[dst]+q[src]) * v[src] * (mask[src]&mask[dst])
// k/q/v stored bf16 (halves the gather working set: 192MB ~= L2 capacity).
__global__ void k_msg(const unsigned short* __restrict__ kb,
                      const unsigned short* __restrict__ qb,
                      const unsigned short* __restrict__ vb,
                      const int* __restrict__ src, const int* __restrict__ dst,
                      const unsigned char* __restrict__ mask,
                      float* __restrict__ out) {
  unsigned int t = blockIdx.x * blockDim.x + threadIdx.x;  // EN*128 threads
  int e = t >> 7;
  int f = (t & 127) << 2;
  int s = src[e], d = dst[e];
  if (!(mask[s] & mask[d])) return;
  uint2 ku = *(const uint2*)(kb + (size_t)d * FH + f);
  uint2 qu = *(const uint2*)(qb + (size_t)s * FH + f);
  uint2 vu = *(const uint2*)(vb + (size_t)s * FH + f);
  float kf[4] = {bf2f(ku.x & 0xffffu), bf2f(ku.x >> 16), bf2f(ku.y & 0xffffu), bf2f(ku.y >> 16)};
  float qf[4] = {bf2f(qu.x & 0xffffu), bf2f(qu.x >> 16), bf2f(qu.y & 0xffffu), bf2f(qu.y >> 16)};
  float vf[4] = {bf2f(vu.x & 0xffffu), bf2f(vu.x >> 16), bf2f(vu.y & 0xffffu), bf2f(vu.y >> 16)};
  float* op = out + (size_t)d * FH + f;
#pragma unroll
  for (int c2 = 0; c2 < 4; ++c2) {
    float g  = kf[c2] + qf[c2];
    float sg = 1.f / (1.f + __expf(-g));
    __hip_atomic_fetch_add(&op[c2], sg * vf[c2], __ATOMIC_RELAXED,
                           __HIP_MEMORY_SCOPE_AGENT);
  }
}

__global__ void k_relu(float* a, int n) {
  int i = blockIdx.x * blockDim.x + threadIdx.x;
  if (i < n) { float v = a[i]; a[i] = v > 0.f ? v : 0.f; }
}

__global__ void k_norm(const float* __restrict__ w, float* __restrict__ out) {
  __shared__ float s[512];
  int t = threadIdx.x;
  float v = w[t];
  s[t] = v * v;
  __syncthreads();
  for (int off = 256; off; off >>= 1) {
    if (t < off) s[t] += s[t + off];
    __syncthreads();
  }
  if (t == 0) out[0] = sqrtf(s[0]);
}

// score[n] = tanh( dot(h[n], w) / ||w|| ) ; one wave per node
__global__ void k_score(const float* __restrict__ h, const float* __restrict__ w,
                        const float* __restrict__ nrm, float* __restrict__ score) {
  int wave = threadIdx.x >> 5;
  int lane = threadIdx.x & 31;
  int node = blockIdx.x * 8 + wave;
  const float* hp = h + (size_t)node * FH;
  float acc = 0.f;
#pragma unroll
  for (int j = 0; j < 16; ++j) {
    int idx = (j << 5) + lane;
    acc += hp[idx] * w[idx];
  }
#pragma unroll
  for (int off = 16; off; off >>= 1) acc += __shfl_xor(acc, off);
  if (lane == 0) score[node] = tanhf(acc / nrm[0]);
}

// Deterministic per-graph top-k: binary-search threshold + index-ordered tie fill.
__global__ void __launch_bounds__(1024) k_topk(const float* __restrict__ score,
                                               const unsigned char* __restrict__ maskin,
                                               unsigned char* __restrict__ maskout, int kk) {
  __shared__ int cnt;
  __shared__ int sb[NG];
  int b = blockIdx.x, t = threadIdx.x;
  int node = b * NG + t;
  float sc = maskin[node] ? score[node] : -2.0f;  // tanh in (-1,1); -2 acts as -inf
  float lo = -3.f, hi = 2.f;
  for (int it = 0; it < 40; ++it) {
    float mid = 0.5f * (lo + hi);
    if (t == 0) cnt = 0;
    __syncthreads();
    if (sc > mid) atomicAdd(&cnt, 1);
    __syncthreads();
    int c = cnt;
    __syncthreads();
    if (c > kk) lo = mid; else hi = mid;
  }
  if (t == 0) cnt = 0;
  __syncthreads();
  if (sc > hi) atomicAdd(&cnt, 1);
  __syncthreads();
  int cgt = cnt;  // count strictly above hi, <= kk
  __syncthreads();
  int flag = (sc > lo && sc <= hi) ? 1 : 0;
  sb[t] = flag;
  __syncthreads();
  for (int off = 1; off < NG; off <<= 1) {      // Hillis-Steele inclusive scan
    int v = (t >= off) ? sb[t - off] : 0;
    __syncthreads();
    sb[t] += v;
    __syncthreads();
  }
  int rank = sb[t] - flag;                       // exclusive rank by node index
  maskout[node] = (unsigned char)((sc > hi) || (flag && (cgt + rank) < kk));
}

__global__ void k_scale(float* __restrict__ h, const float* __restrict__ score) {
  int i = blockIdx.x * blockDim.x + threadIdx.x;
  h[i] *= score[i >> 9];  // FH == 512
}

// z[b,:] += masked mean over graph b's surviving nodes
__global__ void k_gap(const float* __restrict__ h, const unsigned char* __restrict__ mask,
                      float* __restrict__ z) {
  __shared__ float cs[256];
  int b = blockIdx.x, t = threadIdx.x;
  float c = 0.f;
  for (int i = t; i < NG; i += 256) c += (float)mask[b * NG + i];
  cs[t] = c;
  __syncthreads();
  for (int off = 128; off; off >>= 1) {
    if (t < off) cs[t] += cs[t + off];
    __syncthreads();
  }
  float cnt = cs[0];
  for (int f = t; f < FH; f += 256) {
    float s = 0.f;
    for (int node = 0; node < NG; ++node) {
      int idx = b * NG + node;
      s += h[(size_t)idx * FH + f] * (float)mask[idx];
    }
    z[b * FH + f] += s / cnt;
  }
}

__global__ void k_mlp(const float* __restrict__ in, const float* __restrict__ W,
                      const float* __restrict__ b, float* __restrict__ out,
                      int Fin, int Fout, int relu) {
  int t = blockIdx.x * blockDim.x + threadIdx.x;
  if (t >= BG * Fout) return;
  int bb = t / Fout, o = t - bb * Fout;
  float s = b[o];
  for (int i = 0; i < Fin; ++i) s += in[bb * Fin + i] * W[i * Fout + o];
  out[t] = (relu && s < 0.f) ? 0.f : s;
}

extern "C" void kernel_launch(void* const* d_in, const int* in_sizes, int n_in,
                              void* d_out, int out_size, void* d_ws, size_t ws_size,
                              hipStream_t stream) {
  (void)in_sizes; (void)n_in; (void)out_size; (void)ws_size;
  const float* x  = (const float*)d_in[0];
  const int*   ei = (const int*)d_in[1];
  const int* src = ei;
  const int* dst = ei + EN;

  // workspace bump allocator
  char* p = (char*)d_ws;
  auto take = [&](size_t bytes) -> void* {
    void* r = (void*)p;
    p += (bytes + 255) & ~(size_t)255;
    return r;
  };
  unsigned short* hbf = (unsigned short*)take((size_t)MN * FH * 2);
  unsigned short* pk  = (unsigned short*)take((size_t)FH * FH * 2);
  unsigned short* pq  = (unsigned short*)take((size_t)FH * FH * 2);
  unsigned short* pv  = (unsigned short*)take((size_t)FH * FH * 2);
  unsigned short* ps  = (unsigned short*)take((size_t)FH * FH * 2);
  unsigned short* kb  = (unsigned short*)take((size_t)MN * FH * 2);  // bf16 k
  unsigned short* qb  = (unsigned short*)take((size_t)MN * FH * 2);  // bf16 q
  unsigned short* vb  = (unsigned short*)take((size_t)MN * FH * 2);  // bf16 v
  float* h     = (float*)take((size_t)MN * FH * 4);
  float* score = (float*)take((size_t)MN * 4);
  unsigned char* mA = (unsigned char*)take(MN);
  unsigned char* mB = (unsigned char*)take(MN);
  float* nrm = (float*)take(256);
  float* z   = (float*)take((size_t)BG * FH * 4);
  float* z2  = (float*)take((size_t)BG * 256 * 4);
  float* z3  = (float*)take((size_t)BG * 64 * 4);

  k_fill_mask<<<(MN + 255) / 256, 256, 0, stream>>>(mA, MN);
  k_zero<<<(BG * FH + 255) / 256, 256, 0, stream>>>(z, BG * FH);

  unsigned char* mcur = mA;
  unsigned char* mnext = mB;
  const int base[3] = {3, 12, 21};
  const int fins[3] = {FIN0, FH, FH};
  const int kks[3]  = {KK1, KK2, KK3};

  for (int li = 0; li < 3; ++li) {
    int fin = fins[li];
    int bi  = base[li];
    const float* Wk = (const float*)d_in[bi + 0];
    const float* bk = (const float*)d_in[bi + 1];
    const float* Wq = (const float*)d_in[bi + 2];
    const float* bq = (const float*)d_in[bi + 3];
    const float* Wv = (const float*)d_in[bi + 4];
    const float* bv = (const float*)d_in[bi + 5];
    const float* Ws = (const float*)d_in[bi + 6];
    const float* bc = (const float*)d_in[bi + 7];
    const float* pw = (const float*)d_in[bi + 8];
    const float* hin = (li == 0) ? x : h;

    int ne = MN * fin;
    k_f2bf<<<(ne + 255) / 256, 256, 0, stream>>>(hin, hbf, ne);
    int np = fin * FH;
    k_pack_w<<<(np + 255) / 256, 256, 0, stream>>>(Wk, pk, fin, FH);
    k_pack_w<<<(np + 255) / 256, 256, 0, stream>>>(Wq, pq, fin, FH);
    k_pack_w<<<(np + 255) / 256, 256, 0, stream>>>(Wv, pv, fin, FH);
    k_pack_w<<<(np + 255) / 256, 256, 0, stream>>>(Ws, ps, fin, FH);

    int tiles = (MN / 16) * (FH / 64);   // 32768 strips of 16x64
    int gb = tiles / 8;                  // 8 waves per 256-thread block
    k_gemm<<<gb, 256, 0, stream>>>(hbf, pk, bk, nullptr, kb, MN, fin, FH, 1);
    k_gemm<<<gb, 256, 0, stream>>>(hbf, pq, bq, nullptr, qb, MN, fin, FH, 1);
    k_gemm<<<gb, 256, 0, stream>>>(hbf, pv, bv, nullptr, vb, MN, fin, FH, 1);
    k_gemm<<<gb, 256, 0, stream>>>(hbf, ps, bc, h, nullptr, MN, fin, FH, 0);  // skip+bias

    k_msg<<<(EN * 128) / 256, 256, 0, stream>>>(kb, qb, vb, src, dst, mcur, h);
    k_relu<<<(MN * FH) / 256, 256, 0, stream>>>(h, MN * FH);

    k_norm<<<1, 512, 0, stream>>>(pw, nrm);
    k_score<<<MN / 8, 256, 0, stream>>>(h, pw, nrm, score);
    k_topk<<<BG, 1024, 0, stream>>>(score, mcur, mnext, kks[li]);
    k_scale<<<(MN * FH) / 256, 256, 0, stream>>>(h, score);
    k_gap<<<BG, 256, 0, stream>>>(h, mnext, z);

    unsigned char* tmp = mcur; mcur = mnext; mnext = tmp;
  }

  const float* Wl1 = (const float*)d_in[30];
  const float* bl1 = (const float*)d_in[31];
  const float* Wl2 = (const float*)d_in[32];
  const float* bl2 = (const float*)d_in[33];
  const float* Wl3 = (const float*)d_in[34];
  const float* bl3 = (const float*)d_in[35];
  k_mlp<<<(BG * 256 + 255) / 256, 256, 0, stream>>>(z, Wl1, bl1, z2, FH, 256, 1);
  k_mlp<<<(BG * 64 + 255) / 256, 256, 0, stream>>>(z2, Wl2, bl2, z3, 256, 64, 1);
  k_mlp<<<1, 128, 0, stream>>>(z3, Wl3, bl3, (float*)d_out, 64, 2, 0);
}
